// SGAT_multi_75488345194751
// MI455X (gfx1250) — matrix operand order repaired and checked by reading, compiled, not run on hardware
//
#include <hip/hip_runtime.h>
#include <hip/hip_bf16.h>

typedef _Float16 half_t;
typedef __attribute__((ext_vector_type(16))) _Float16 v16h;
typedef __attribute__((ext_vector_type(8)))  _Float16 v8h;
typedef __attribute__((ext_vector_type(8)))  float    v8f;

#define NFEAT  128
#define NHID   64
#define NHEAD  4
#define NBASE  8
#define NHEAD2 1
#define NCLASS 40
#define NEG_SLOPE 0.2f

// ---------------------------------------------------------------- utilities

__global__ void k_fill(float* __restrict__ p, float v, long n) {
    long t = (long)blockIdx.x * blockDim.x + threadIdx.x;
    if (t < n) p[t] = v;
}

__global__ void k_cvt_f16(const float* __restrict__ s, half_t* __restrict__ d, long n) {
    long t = (long)blockIdx.x * blockDim.x + threadIdx.x;
    if (t < n) d[t] = (half_t)s[t];
}

// W [K][Ncols] row-major f32  ->  Wt [Npad][K] f16 (K-contiguous per output col, zero pad)
__global__ void k_prep_bt(const float* __restrict__ W, half_t* __restrict__ Wt,
                          int K, int Ncols, int Npad) {
    int t = blockIdx.x * blockDim.x + threadIdx.x;
    if (t >= Npad * K) return;
    int n = t / K, k = t - n * K;
    float v = (n < Ncols) ? W[k * Ncols + n] : 0.0f;
    Wt[(long)n * K + k] = (half_t)v;
}

// ------------------------------------------------------------- WMMA GEMM
// C[M][Ncols] (f32) = A[M][K] (f16, row-major) x Bt[tilesN*16][K] (f16, K-major per col)
// One wave computes a 16 x (NB*16) strip: one A fragment is reused by NB WMMAs,
// raising FLOPs/byte NB-fold. K stepped by 32 via v_wmma_f32_16x16x32_f16.
// 16-bit A 16x32 layout: lane m=lane&15, hi=lane>>4; elems 0..7 = K(8*hi+0..7),
//                        elems 8..15 = K(16+8*hi+0..7).
// 16-bit B 32x16 layout: lane n=lane&15, hi=lane>>4; elems 0..7 = K(8*hi+0..7), 8..15 = K(16+8*hi+0..7).
// C/D: VGPR v -> row 8*hi+v, col lane&15.
template<int NB>
__global__ void k_wmma_gemm(const half_t* __restrict__ A, const half_t* __restrict__ Bt,
                            float* __restrict__ C, int M, int K, int Ncols, int tilesN) {
    const int groups = tilesN / NB;              // N-tile groups per row strip
    int wave = (int)((blockIdx.x * (long)blockDim.x + threadIdx.x) >> 5);
    int lane = threadIdx.x & 31;
    int tm = wave / groups;
    int g  = wave - tm * groups;
    if (tm * 16 >= M) return;

    int hi = lane >> 4;
    int m  = tm * 16 + (lane & 15);

    const half_t* ap  = A  + (long)m * K + 8 * hi;
    const half_t* bp0 = Bt + (long)(g * NB * 16 + (lane & 15)) * K + 8 * hi;

    v8f acc[NB];
    #pragma unroll
    for (int j = 0; j < NB; ++j) acc[j] = (v8f){};

    for (int k0 = 0; k0 < K; k0 += 32) {
        union { v16h v; v8h h2[2]; } a;
        a.h2[0] = *(const v8h*)(ap + k0);
        a.h2[1] = *(const v8h*)(ap + k0 + 16);
        #pragma unroll
        for (int j = 0; j < NB; ++j) {
            union { v16h v; v8h h2[2]; } b;
            const half_t* bp = bp0 + (long)j * 16 * K;
            b.h2[0] = *(const v8h*)(bp + k0);
            b.h2[1] = *(const v8h*)(bp + k0 + 16);
            acc[j] = __builtin_amdgcn_wmma_f32_16x16x32_f16(
                false, a.v, false, b.v, (short)0, acc[j], false, false);
        }
    }

    long rb = (long)(tm * 16 + 8 * hi);
    #pragma unroll
    for (int j = 0; j < NB; ++j) {
        int col = (g * NB + j) * 16 + (lane & 15);
        if (col < Ncols) {
            #pragma unroll
            for (int v = 0; v < 8; ++v)
                C[(rb + v) * Ncols + col] = acc[j][v];
        }
    }
}

// ------------------------------------------- per-node separable attention logits
// s = X[node] @ Bm (Fin x 8), es = s @ cs (8 x nhead), ed = s @ cd
__global__ void k_node_attn(const float* __restrict__ X, const float* __restrict__ Bm,
                            const float* __restrict__ cs, const float* __restrict__ cd,
                            float* __restrict__ es, float* __restrict__ ed,
                            int N, int Fin, int nhead) {
    int node = blockIdx.x * blockDim.x + threadIdx.x;
    if (node >= N) return;
    float s[NBASE];
    #pragma unroll
    for (int b = 0; b < NBASE; ++b) s[b] = 0.0f;
    const float* xr = X + (long)node * Fin;
    #pragma unroll 4
    for (int k = 0; k < Fin; ++k) {
        float xv = xr[k];
        #pragma unroll
        for (int b = 0; b < NBASE; ++b)
            s[b] = fmaf(xv, Bm[k * NBASE + b], s[b]);
    }
    for (int h = 0; h < nhead; ++h) {
        float a = 0.0f, c = 0.0f;
        #pragma unroll
        for (int b = 0; b < NBASE; ++b) {
            a = fmaf(s[b], cs[b * nhead + h], a);
            c = fmaf(s[b], cd[b * nhead + h], c);
        }
        es[(long)node * nhead + h] = a;
        ed[(long)node * nhead + h] = c;
    }
}

__device__ __forceinline__ float leaky(float v) {
    return v > 0.0f ? v : NEG_SLOPE * v;
}

// exact float atomic max (signed-int for >=0, unsigned-min for <0; init = -inf)
__device__ __forceinline__ void atomicMaxF(float* addr, float v) {
    if (v >= 0.0f) atomicMax((int*)addr, __float_as_int(v));
    else           atomicMin((unsigned int*)addr, __float_as_uint(v));
}

// ----------------------------------------------------- edge softmax phases
__global__ void k_edge_max(const int* __restrict__ src, const int* __restrict__ dst,
                           const float* __restrict__ es, const float* __restrict__ ed,
                           float* __restrict__ m, long E, int nhead) {
    long t = (long)blockIdx.x * blockDim.x + threadIdx.x;
    if (t >= E * nhead) return;
    long e = t / nhead; int h = (int)(t - e * nhead);
    int s = src[e], d = dst[e];
    float v = leaky(es[(long)s * nhead + h] + ed[(long)d * nhead + h]);
    atomicMaxF(m + (long)d * nhead + h, v);
}

__global__ void k_edge_denom(const int* __restrict__ src, const int* __restrict__ dst,
                             const float* __restrict__ es, const float* __restrict__ ed,
                             const float* __restrict__ m, float* __restrict__ den,
                             long E, int nhead) {
    long t = (long)blockIdx.x * blockDim.x + threadIdx.x;
    if (t >= E * nhead) return;
    long e = t / nhead; int h = (int)(t - e * nhead);
    int s = src[e], d = dst[e];
    float v  = leaky(es[(long)s * nhead + h] + ed[(long)d * nhead + h]);
    float ee = __expf(v - m[(long)d * nhead + h]);
    atomicAdd(den + (long)d * nhead + h, ee);
}

// materialize softmax weights alpha[E*nhead] once (keeps exp/divide and the
// 4-scalar gather out of the 51M-thread scatter kernel)
__global__ void k_edge_alpha(const int* __restrict__ src, const int* __restrict__ dst,
                             const float* __restrict__ es, const float* __restrict__ ed,
                             const float* __restrict__ m, const float* __restrict__ den,
                             float* __restrict__ alpha, long E, int nhead) {
    long t = (long)blockIdx.x * blockDim.x + threadIdx.x;
    if (t >= E * nhead) return;
    long e = t / nhead; int h = (int)(t - e * nhead);
    int s = src[e], d = dst[e];
    float v = leaky(es[(long)s * nhead + h] + ed[(long)d * nhead + h]);
    alpha[t] = __expf(v - m[(long)d * nhead + h]) /
               (den[(long)d * nhead + h] + 1e-16f);
}

// weighted scatter: out[dst] += alpha * H[src], 4 floats per thread.
// threads_per_edge = nhead*D/4 ; D multiple of 4, rows 16B-aligned.
// out accumulators are L2-resident (<=52MB << 192MB) so atomics resolve in L2.
__global__ void k_edge_scatter(const int* __restrict__ src, const int* __restrict__ dst,
                               const float* __restrict__ alpha,
                               const float* __restrict__ H, float* __restrict__ out,
                               long E, int nhead, int D) {
    int dq  = D >> 2;            // float4 chunks per head
    int tpe = nhead * dq;        // threads per edge
    long t = (long)blockIdx.x * blockDim.x + threadIdx.x;
    long eid = t / tpe;
    if (eid >= E) return;
    int r = (int)(t - eid * tpe);
    int h = r / dq;
    int c = (r - h * dq) << 2;
    int s = src[eid], d = dst[eid];
    float a = alpha[eid * nhead + h];
    const float4 hv = *(const float4*)(H + (long)s * nhead * D + h * D + c);
    float* o = out + (long)d * nhead * D + h * D + c;
    atomicAdd(o + 0, a * hv.x);
    atomicAdd(o + 1, a * hv.y);
    atomicAdd(o + 2, a * hv.z);
    atomicAdd(o + 3, a * hv.w);
}

// elu in place + f16 copy for next WMMA stage
__global__ void k_elu_cvt(float* __restrict__ io, const float* __restrict__ bias,
                          half_t* __restrict__ oh, long n, int cols) {
    long t = (long)blockIdx.x * blockDim.x + threadIdx.x;
    if (t >= n) return;
    float v = io[t] + bias[t % cols];
    v = v > 0.0f ? v : (__expf(v) - 1.0f);
    io[t] = v;
    oh[t] = (half_t)v;
}

// ---------------------------------------------------------------- launcher

static inline int ceil_div_l(long a, int b) { return (int)((a + b - 1) / b); }

extern "C" void kernel_launch(void* const* d_in, const int* in_sizes, int n_in,
                              void* d_out, int out_size, void* d_ws, size_t ws_size,
                              hipStream_t stream) {
    const float* x   = (const float*)d_in[0];
    const int*   ei  = (const int*)d_in[1];
    const float* W1  = (const float*)d_in[2];
    const float* b1  = (const float*)d_in[3];
    const float* B1  = (const float*)d_in[4];
    const float* cs1 = (const float*)d_in[5];
    const float* cd1 = (const float*)d_in[6];
    const float* W2  = (const float*)d_in[7];
    const float* B2  = (const float*)d_in[8];
    const float* cs2 = (const float*)d_in[9];
    const float* cd2 = (const float*)d_in[10];
    float* out = (float*)d_out;

    const long N = in_sizes[0] / NFEAT;     // 50000
    const long E = in_sizes[1] / 2;         // 800000
    const int  F1 = NFEAT;                  // 128
    const int  C1 = NHEAD * NHID;           // 256
    const int  C2 = NHEAD2 * NCLASS;        // 40
    const int  C2PAD = 48;                  // 3 N-tiles of 16

    const int* srcI = ei;
    const int* dstI = ei + E;

    // ----- workspace partition (256B aligned slabs) -----
    char* wp = (char*)d_ws;
    size_t off = 0;
    auto alloc = [&](size_t bytes) -> void* {
        void* p = wp + off;
        off += (bytes + 255) & ~(size_t)255;
        return p;
    };
    half_t* xh   = (half_t*)alloc((size_t)N * F1 * sizeof(half_t));
    half_t* W1t  = (half_t*)alloc((size_t)C1 * F1 * sizeof(half_t));
    half_t* W2t  = (half_t*)alloc((size_t)C2PAD * C1 * sizeof(half_t));
    float*  h1   = (float*)alloc((size_t)N * C1 * sizeof(float));
    float*  out1 = (float*)alloc((size_t)N * C1 * sizeof(float));   // becomes h (post-ELU)
    half_t* hh   = (half_t*)alloc((size_t)N * C1 * sizeof(half_t));
    float*  h2   = (float*)alloc((size_t)N * C2 * sizeof(float));
    float*  al1  = (float*)alloc((size_t)E * NHEAD * sizeof(float));
    float*  al2  = (float*)alloc((size_t)E * sizeof(float));
    float*  es1  = (float*)alloc((size_t)N * NHEAD * sizeof(float));
    float*  ed1  = (float*)alloc((size_t)N * NHEAD * sizeof(float));
    float*  m1   = (float*)alloc((size_t)N * NHEAD * sizeof(float));
    float*  den1 = (float*)alloc((size_t)N * NHEAD * sizeof(float));
    float*  es2  = (float*)alloc((size_t)N * sizeof(float));
    float*  ed2  = (float*)alloc((size_t)N * sizeof(float));
    float*  m2   = (float*)alloc((size_t)N * sizeof(float));
    float*  den2 = (float*)alloc((size_t)N * sizeof(float));
    (void)ws_size;

    const int B = 256;

    // ----- init accumulators -----
    k_fill<<<ceil_div_l(N * C1, B), B, 0, stream>>>(out1, 0.0f, N * C1);
    k_fill<<<ceil_div_l(N * NHEAD, B), B, 0, stream>>>(den1, 0.0f, N * NHEAD);
    k_fill<<<ceil_div_l(N * NHEAD, B), B, 0, stream>>>(m1, -INFINITY, N * NHEAD);
    k_fill<<<ceil_div_l(N, B), B, 0, stream>>>(den2, 0.0f, N);
    k_fill<<<ceil_div_l(N, B), B, 0, stream>>>(m2, -INFINITY, N);
    k_fill<<<ceil_div_l((long)N * C2, B), B, 0, stream>>>(out, 0.0f, (long)N * C2);

    // ----- f16 staging -----
    k_cvt_f16<<<ceil_div_l(N * F1, B), B, 0, stream>>>(x, xh, N * F1);
    k_prep_bt<<<ceil_div_l((long)C1 * F1, B), B, 0, stream>>>(W1, W1t, F1, C1, C1);
    k_prep_bt<<<ceil_div_l((long)C2PAD * C1, B), B, 0, stream>>>(W2, W2t, C1, C2, C2PAD);

    // ----- layer 1 -----
    {
        long waves = (N / 16) * ((C1 / 16) / 4);    // NB=4
        k_wmma_gemm<4><<<ceil_div_l(waves * 32, B), B, 0, stream>>>(
            xh, W1t, h1, (int)N, F1, C1, C1 / 16);
    }
    k_node_attn<<<ceil_div_l(N, B), B, 0, stream>>>(x, B1, cs1, cd1, es1, ed1,
                                                    (int)N, F1, NHEAD);
    k_edge_max<<<ceil_div_l(E * NHEAD, B), B, 0, stream>>>(srcI, dstI, es1, ed1, m1, E, NHEAD);
    k_edge_denom<<<ceil_div_l(E * NHEAD, B), B, 0, stream>>>(srcI, dstI, es1, ed1, m1, den1,
                                                             E, NHEAD);
    k_edge_alpha<<<ceil_div_l(E * NHEAD, B), B, 0, stream>>>(srcI, dstI, es1, ed1, m1, den1,
                                                             al1, E, NHEAD);
    k_edge_scatter<<<ceil_div_l(E * (NHEAD * NHID / 4), B), B, 0, stream>>>(
        srcI, dstI, al1, h1, out1, E, NHEAD, NHID);
    k_elu_cvt<<<ceil_div_l(N * C1, B), B, 0, stream>>>(out1, b1, hh, N * C1, C1);

    // ----- layer 2 -----
    {
        long waves = (N / 16) * ((C2PAD / 16) / 3); // NB=3, one group
        k_wmma_gemm<3><<<ceil_div_l(waves * 32, B), B, 0, stream>>>(
            hh, W2t, h2, (int)N, C1, C2, C2PAD / 16);
    }
    k_node_attn<<<ceil_div_l(N, B), B, 0, stream>>>(out1, B2, cs2, cd2, es2, ed2,
                                                    (int)N, C1, NHEAD2);
    k_edge_max<<<ceil_div_l(E, B), B, 0, stream>>>(srcI, dstI, es2, ed2, m2, E, NHEAD2);
    k_edge_denom<<<ceil_div_l(E, B), B, 0, stream>>>(srcI, dstI, es2, ed2, m2, den2, E, NHEAD2);
    k_edge_alpha<<<ceil_div_l(E, B), B, 0, stream>>>(srcI, dstI, es2, ed2, m2, den2,
                                                     al2, E, NHEAD2);
    k_edge_scatter<<<ceil_div_l(E * (NCLASS / 4), B), B, 0, stream>>>(
        srcI, dstI, al2, h2, out, E, NHEAD2, NCLASS);
}
